// KEENKT_79353815760943
// MI455X (gfx1250) — matrix operand order repaired and checked by reading, not compiled, not run on hardware
//
#include <hip/hip_runtime.h>
#include <hip/hip_bf16.h>

// ---------------------------------------------------------------------------
// Problem constants (reference: B=4, S=1024, D=512, H=8, DK=64)
// ---------------------------------------------------------------------------
constexpr int cB  = 4;
constexpr int cS  = 1024;
constexpr int cD  = 512;
constexpr int cH  = 8;
constexpr int cDK = 64;

typedef __attribute__((ext_vector_type(16))) _Float16 v16h;
typedef __attribute__((ext_vector_type(8)))  _Float16 v8h;
typedef __attribute__((ext_vector_type(8)))  float    v8f;
typedef __attribute__((ext_vector_type(4)))  int      v4i;

static __device__ __forceinline__ v8f wmma_f16(v16h a, v16h b, v8f c) {
    // v_wmma_f32_16x16x32_f16 : D = A(16x32) * B(32x16) + C(16x16 f32)
    return __builtin_amdgcn_wmma_f32_16x16x32_f16(
        /*neg_a=*/false, a, /*neg_b=*/false, b,
        /*c_mod=*/(short)0, c, /*reuse_a=*/false, /*reuse_b=*/false);
}

// ---------------------------------------------------------------------------
// gfx1250 async global->LDS copy (ASYNCcnt-tracked), guarded so the file
// still compiles if this toolchain does not declare the builtins.
// Probe-confirmed signature: (global int4*, lds int4*, imm offset, imm cpol)
// ---------------------------------------------------------------------------
#if defined(__has_builtin)
#if __has_builtin(__builtin_amdgcn_global_load_async_to_lds_b128)
#define ASYNC_LDS 1
#endif
#endif

#if defined(ASYNC_LDS)
typedef __attribute__((address_space(1))) v4i g_v4i;  // global int4
typedef __attribute__((address_space(3))) v4i l_v4i;  // LDS int4
static __device__ __forceinline__ void async_copy_b128(const void* g, void* l) {
    __builtin_amdgcn_global_load_async_to_lds_b128(
        (g_v4i*)g, (l_v4i*)l, 0, 0);
}
#endif

static __device__ __forceinline__ void wait_async_lds() {
#if defined(ASYNC_LDS)
#if __has_builtin(__builtin_amdgcn_s_wait_asynccnt)
    __builtin_amdgcn_s_wait_asynccnt(0);
#else
    asm volatile("s_wait_asynccnt 0x0" ::: "memory");
#endif
#endif
}

// ---------------------------------------------------------------------------
// fp32 -> f16 conversion
// ---------------------------------------------------------------------------
__global__ void cvt_f32_f16(const float* __restrict__ in,
                            _Float16* __restrict__ out, int n) {
    int i = blockIdx.x * blockDim.x + threadIdx.x;
    if (i < n) out[i] = (_Float16)in[i];
}

// ---------------------------------------------------------------------------
// GEMM: C[M,N] = A[M,K](f16) * W[K,N](f16) + bias[N](f32)
// Block = 128 threads (4 waves), tile 64(M) x 64(N), K-step 64.
// Each wave computes a 16x64 strip (4 WMMA accumulators, 8 WMMA per stage).
// A tile staged with async global->LDS; W tile staged transposed.
// Output either f32 (Cf) or f16 (Ch).
// ---------------------------------------------------------------------------
__global__ __launch_bounds__(128) void gemm_f16_wmma(
    const _Float16* __restrict__ A, const _Float16* __restrict__ W,
    const float* __restrict__ bias, float* __restrict__ Cf,
    _Float16* __restrict__ Ch, int M, int N, int K) {
    __shared__ __align__(32) _Float16 sA[64 * 64];   // [m][k] row-major
    __shared__ __align__(32) _Float16 sBT[64 * 64];  // [n][k] (W transposed)

    const int tid  = threadIdx.x;
    const int lane = tid & 31;
    const int w    = tid >> 5;
    const int bm   = blockIdx.y * 64;
    const int bn   = blockIdx.x * 64;

    v8f acc[4];
#pragma unroll
    for (int c = 0; c < 4; ++c)
#pragma unroll
        for (int v = 0; v < 8; ++v) acc[c][v] = 0.f;

    const int mloc = w * 16 + (lane & 15);
    const int kb   = (lane >> 4) * 16;  // A fragment K sub-base
    const int hk   = (lane >> 4) * 16;  // B fragment K sub-base

    for (int k0 = 0; k0 < K; k0 += 64) {
        // ---- stage A 64x64 tile ------------------------------------------
#if defined(ASYNC_LDS)
        // 16B (8 x f16) per async op: 512 chunks, 4 per thread
        for (int idx = tid; idx < 512; idx += 128) {
            int i  = idx >> 3;        // row 0..63
            int c8 = (idx & 7) * 8;   // col chunk
            async_copy_b128((const void*)&A[(size_t)(bm + i) * K + k0 + c8],
                            (void*)&sA[i * 64 + c8]);
        }
#else
        for (int idx = tid; idx < 64 * 64; idx += 128) {
            int i = idx >> 6, kk = idx & 63;
            sA[idx] = A[(size_t)(bm + i) * K + k0 + kk];
        }
#endif
        // ---- stage W 64x64 tile, transposed -> sBT[n][k] -----------------
        for (int idx = tid; idx < 64 * 64; idx += 128) {
            int kl = idx >> 6, nl = idx & 63;
            sBT[nl * 64 + kl] = W[(size_t)(k0 + kl) * N + bn + nl];
        }
        // ---- prefetch next stage into L2/L0 ------------------------------
        if (k0 + 64 < K) {
            int t2 = tid & 63;
            if (tid < 64)
                __builtin_prefetch(&A[(size_t)(bm + t2) * K + k0 + 64], 0, 1);
            else
                __builtin_prefetch(&W[(size_t)(k0 + 64 + t2) * N + bn], 0, 1);
        }
        wait_async_lds();
        __syncthreads();

        // ---- 2 depth chunks x 4 column tiles = 8 WMMA --------------------
#pragma unroll
        for (int d = 0; d < 2; ++d) {
            const int d0 = d * 32;
            v8h lo = *(const v8h*)&sA[mloc * 64 + d0 + kb];
            v8h hi = *(const v8h*)&sA[mloc * 64 + d0 + kb + 8];
            v16h af;
#pragma unroll
            for (int j = 0; j < 8; ++j) { af[j] = lo[j]; af[j + 8] = hi[j]; }
#pragma unroll
            for (int c = 0; c < 4; ++c) {
                int nloc = c * 16 + (lane & 15);
                v16h bf = *(const v16h*)&sBT[nloc * 64 + d0 + hk];
                acc[c] = wmma_f16(af, bf, acc[c]);
            }
        }
        __syncthreads();
    }

    // C layout: lane&15 = N, vgpr v -> M = v + 8*(lane>>4)
#pragma unroll
    for (int c = 0; c < 4; ++c) {
        int n = bn + c * 16 + (lane & 15);
        float bv = bias[n];
#pragma unroll
        for (int v = 0; v < 8; ++v) {
            int m = bm + w * 16 + v + 8 * (lane >> 4);
            float val = acc[c][v] + bv;
            if (Cf) Cf[(size_t)m * N + n] = val;
            else    Ch[(size_t)m * N + n] = (_Float16)val;
        }
    }
}

// ---------------------------------------------------------------------------
// Fused monotonic attention for one (b, h, 16-query tile).
// Grid: (S/16, B*H). Block: 128 threads (4 waves).
// Dynamic LDS: sS[16*1024] f32 | sVT[64*64] f16 | sRed[16*8] f32
// ---------------------------------------------------------------------------
__global__ __launch_bounds__(128) void attn_f16_wmma(
    const _Float16* __restrict__ Q, const _Float16* __restrict__ Kt,
    const _Float16* __restrict__ V, _Float16* __restrict__ Out,
    const float* __restrict__ gammas) {
    extern __shared__ char smem[];
    float*    sS   = (float*)smem;                               // 65536 B
    _Float16* sVT  = (_Float16*)(smem + 16 * 1024 * 4);          //  8192 B
    float*    sRed = (float*)(smem + 16 * 1024 * 4 + 64 * 64 * 2); // 512 B

    const int tid  = threadIdx.x;
    const int lane = tid & 31;
    const int w    = tid >> 5;
    const int q0   = blockIdx.x * 16;
    const int b    = blockIdx.y / cH;
    const int h    = blockIdx.y % cH;

    const int mq = lane & 15;         // fragment row / C column
    const int kb = (lane >> 4) * 16;  // A fragment K sub-base
    const int hk = (lane >> 4) * 16;  // B fragment K sub-base

    // ---- Phase 1: scores S[16, 1024] = Q_tile @ K^T * (1/sqrt(DK)) --------
    const _Float16* qrow = Q + ((size_t)(b * cS + q0 + mq)) * cD + h * cDK;
    v8h q00 = *(const v8h*)(qrow + kb);
    v8h q01 = *(const v8h*)(qrow + 8 + kb);
    v8h q10 = *(const v8h*)(qrow + 32 + kb);
    v8h q11 = *(const v8h*)(qrow + 40 + kb);
    v16h a0, a1;
#pragma unroll
    for (int j = 0; j < 8; ++j) {
        a0[j] = q00[j]; a0[j + 8] = q01[j];
        a1[j] = q10[j]; a1[j + 8] = q11[j];
    }

    for (int t = w; t < cS / 16; t += 4) {
        int kk = t * 16;
        // B = K^T chunk: lane&15 selects key, 16 contiguous features per lane
        const _Float16* krow =
            Kt + ((size_t)(b * cS + kk + mq)) * cD + h * cDK + hk;
        v16h b0 = *(const v16h*)(krow);
        v16h b1 = *(const v16h*)(krow + 32);
        v8f c;
#pragma unroll
        for (int v = 0; v < 8; ++v) c[v] = 0.f;
        c = wmma_f16(a0, b0, c);
        c = wmma_f16(a1, b1, c);
        int col = kk + mq;
#pragma unroll
        for (int v = 0; v < 8; ++v)
            sS[(v + 8 * (lane >> 4)) * cS + col] = c[v] * 0.125f;
    }
    __syncthreads();

    // ---- Phase 2: per-row softmax -> cumsum -> distance decay -> softmax --
    // 8 threads per row, each owns a 128-wide segment.
    {
        const int r    = tid >> 3;
        const int seg  = tid & 7;
        const int qg   = q0 + r;          // causal: valid k <= qg
        float* row     = sS + r * cS;
        const int kbeg = seg * 128, kend = kbeg + 128;

        float g  = gammas[h];
        float sp = (g > 20.f) ? g : log1pf(__expf(g));
        float gv = -sp;                    // -softplus(gamma)

        // max1
        float mx = -3.0e38f;
        for (int k = kbeg; k < kend; ++k)
            if (k <= qg) mx = fmaxf(mx, row[k]);
        sRed[r * 8 + seg] = mx;
        __syncthreads();
        float mx1 = -3.0e38f;
        for (int i = 0; i < 8; ++i) mx1 = fmaxf(mx1, sRed[r * 8 + i]);
        __syncthreads();

        // segment exp-sums -> total + exclusive prefix (for cumsum)
        float ls = 0.f;
        for (int k = kbeg; k < kend; ++k)
            if (k <= qg) ls += __expf(row[k] - mx1);
        sRed[r * 8 + seg] = ls;
        __syncthreads();
        float sum1 = 0.f, pre = 0.f;
        for (int i = 0; i < 8; ++i) {
            if (i < seg) pre += sRed[r * 8 + i];
            sum1 += sRed[r * 8 + i];
        }
        __syncthreads();
        float inv1 = 1.f / sum1;

        // transform: p -> running cumsum -> d=sqrt((1-cum)*|dpos|) -> s*e
        float run  = pre * inv1;
        float mx2l = -3.0e38f;
        for (int k = kbeg; k < kend; ++k) {
            if (k <= qg) {
                float p = __expf(row[k] - mx1) * inv1;
                run += p;
                float rem = 1.f - run;
                rem = (rem < 0.f) ? 0.f : rem;
                float dd = __fsqrt_rn(rem * (float)(qg - k));
                float e  = __expf(dd * gv);
                e = fminf(fmaxf(e, 1e-5f), 1e5f);
                float l2 = row[k] * e;
                row[k] = l2;
                mx2l = fmaxf(mx2l, l2);
            } else {
                row[k] = -3.0e38f;
            }
        }
        sRed[r * 8 + seg] = mx2l;
        __syncthreads();
        float mx2 = -3.0e38f;
        for (int i = 0; i < 8; ++i) mx2 = fmaxf(mx2, sRed[r * 8 + i]);
        __syncthreads();

        // second softmax sum
        float ls2 = 0.f;
        for (int k = kbeg; k < kend; ++k)
            if (k <= qg) ls2 += __expf(row[k] - mx2);
        sRed[r * 8 + seg] = ls2;
        __syncthreads();
        float sum2 = 0.f;
        for (int i = 0; i < 8; ++i) sum2 += sRed[r * 8 + i];
        float inv2 = 1.f / sum2;

        // normalize (invalid -> 0 so PV matmul is naturally masked)
        for (int k = kbeg; k < kend; ++k)
            row[k] = (k <= qg) ? __expf(row[k] - mx2) * inv2 : 0.f;
    }
    __syncthreads();

    // ---- Phase 3: O[16, 64] = P[16,1024] @ V[1024,64] ---------------------
    const int f0 = w * 16;  // each wave owns 16 output features
    v8f acc;
#pragma unroll
    for (int v = 0; v < 8; ++v) acc[v] = 0.f;

    for (int kc = 0; kc < cS; kc += 64) {
        __syncthreads();
        // stage V chunk transposed: sVT[feature][key], 64 keys
        for (int idx = tid; idx < 64 * 64; idx += 128) {
            int key = idx >> 6, f = idx & 63;
            sVT[f * 64 + key] =
                V[((size_t)(b * cS + kc + key)) * cD + h * cDK + f];
        }
        __syncthreads();

#pragma unroll
        for (int c0 = 0; c0 < 64; c0 += 32) {
            // A fragment = probabilities (f32 LDS -> f16)
            const float* prow = sS + mq * cS + kc + c0;
            v16h af;
#pragma unroll
            for (int j = 0; j < 8; ++j) {
                af[j]     = (_Float16)prow[kb + j];
                af[j + 8] = (_Float16)prow[8 + kb + j];
            }
            v16h bf = *(const v16h*)&sVT[(f0 + mq) * 64 + c0 + hk];
            acc = wmma_f16(af, bf, acc);
        }
    }

    const int fo = h * cDK + f0 + mq;
#pragma unroll
    for (int v = 0; v < 8; ++v) {
        int m = v + 8 * (lane >> 4);
        Out[((size_t)(b * cS + q0 + m)) * cD + fo] = (_Float16)acc[v];
    }
}

// ---------------------------------------------------------------------------
// Host launcher
// ---------------------------------------------------------------------------
extern "C" void kernel_launch(void* const* d_in, const int* in_sizes, int n_in,
                              void* d_out, int out_size, void* d_ws,
                              size_t ws_size, hipStream_t stream) {
    (void)in_sizes; (void)n_in; (void)out_size; (void)ws_size;

    constexpr int ACT = cB * cS * cD;  // 2,097,152
    constexpr int WEL = cD * cD;       //   262,144

    // f16 workspace layout
    _Float16* ws16  = (_Float16*)d_ws;
    _Float16* acts  = ws16;                    // 6 x ACT (converted acts)
    _Float16* wts   = acts + 6 * (size_t)ACT;  // 6 x WEL (converted weights)
    _Float16* proj  = wts + 6 * (size_t)WEL;   // 6 x ACT (qm,qc,km,kc,vm,vc)
    _Float16* attno = proj + 6 * (size_t)ACT;  // 2 x ACT (om, oc)

    // convert activations (inputs 0..5)
    for (int i = 0; i < 6; ++i)
        cvt_f32_f16<<<(ACT + 255) / 256, 256, 0, stream>>>(
            (const float*)d_in[i], acts + (size_t)i * ACT, ACT);
    // convert weights: Wk_mean(6), Wk_cov(8), Wv_mean(10), Wv_cov(12),
    //                  Wo_mean(14), Wo_cov(16)
    const int widx[6] = {6, 8, 10, 12, 14, 16};
    for (int i = 0; i < 6; ++i)
        cvt_f32_f16<<<(WEL + 255) / 256, 256, 0, stream>>>(
            (const float*)d_in[widx[i]], wts + (size_t)i * WEL, WEL);

    const float* bk_mean = (const float*)d_in[7];
    const float* bk_cov  = (const float*)d_in[9];
    const float* bv_mean = (const float*)d_in[11];
    const float* bv_cov  = (const float*)d_in[13];
    const float* bo_mean = (const float*)d_in[15];
    const float* bo_cov  = (const float*)d_in[17];
    const float* gammas  = (const float*)d_in[18];

    _Float16* wk_m = wts + 0 * (size_t)WEL;
    _Float16* wk_c = wts + 1 * (size_t)WEL;
    _Float16* wv_m = wts + 2 * (size_t)WEL;
    _Float16* wv_c = wts + 3 * (size_t)WEL;
    _Float16* wo_m = wts + 4 * (size_t)WEL;
    _Float16* wo_c = wts + 5 * (size_t)WEL;

    const dim3 ggrid(cD / 64, (cB * cS) / 64);  // (8, 64)
    const dim3 gblk(128);

    // projections (kq_same: queries+keys use the K linears)
    struct PJ { int a; _Float16* w; const float* b; int o; };
    PJ pjs[6] = {
        {0, wk_m, bk_mean, 0},  // qm
        {1, wk_c, bk_cov,  1},  // qc
        {2, wk_m, bk_mean, 2},  // km
        {3, wk_c, bk_cov,  3},  // kc
        {4, wv_m, bv_mean, 4},  // vm
        {5, wv_c, bv_cov,  5},  // vc
    };
    for (int i = 0; i < 6; ++i)
        gemm_f16_wmma<<<ggrid, gblk, 0, stream>>>(
            acts + (size_t)pjs[i].a * ACT, pjs[i].w, pjs[i].b,
            nullptr, proj + (size_t)pjs[i].o * ACT,
            cB * cS, cD, cD);

    // fused attention (mean then cov)
    const dim3 agrid(cS / 16, cB * cH);  // (64, 32)
    const size_t asmem = 16 * 1024 * 4 + 64 * 64 * 2 + 16 * 8 * 4;  // 74240 B
    attn_f16_wmma<<<agrid, gblk, asmem, stream>>>(
        proj + 0 * (size_t)ACT, proj + 2 * (size_t)ACT,
        proj + 4 * (size_t)ACT, attno + 0 * (size_t)ACT, gammas);
    attn_f16_wmma<<<agrid, gblk, asmem, stream>>>(
        proj + 1 * (size_t)ACT, proj + 3 * (size_t)ACT,
        proj + 5 * (size_t)ACT, attno + 1 * (size_t)ACT, gammas);

    // output projections, fp32 into d_out (mean first, then cov)
    float* outf = (float*)d_out;
    gemm_f16_wmma<<<ggrid, gblk, 0, stream>>>(
        attno + 0 * (size_t)ACT, wo_m, bo_mean, outf, nullptr,
        cB * cS, cD, cD);
    gemm_f16_wmma<<<ggrid, gblk, 0, stream>>>(
        attno + 1 * (size_t)ACT, wo_c, bo_cov, outf + (size_t)ACT, nullptr,
        cB * cS, cD, cD);
}
